// CustomFlashAttention_1906965479777
// MI455X (gfx1250) — compile-verified, matrix operand order checked by reading, not run on hardware
//
#include <hip/hip_runtime.h>
#include <hip/hip_bf16.h>

// ---------------------------------------------------------------------------
// MHA block for MI455X (gfx1250, wave32, WMMA + TDM).
//   q = x Wq^T ; k = x Wk^T ; v = x Wv^T      (bf16 GEMMs, f32 accum, TDM-staged)
//   o = softmax(q k^T / sqrt(hd)) v            (flash-attention, WMMA)
//   out = o Wo^T                               (bf16 GEMM, f32 out)
// ---------------------------------------------------------------------------

typedef __bf16 bf16;
typedef bf16  v16bf __attribute__((ext_vector_type(16)));
typedef float v8f   __attribute__((ext_vector_type(8)));
typedef unsigned int u32x4 __attribute__((ext_vector_type(4)));
typedef int i32x4 __attribute__((ext_vector_type(4)));
typedef int i32x8 __attribute__((ext_vector_type(8)));

#define BATCH 2
#define SEQ   2048
#define EMB   2048
#define HEADS 16
#define HDIM  128

// ---------------------------------------------------------------------------
// fp32 -> bf16 conversion (4 elems/thread)
// ---------------------------------------------------------------------------
__global__ void cvt_f32_bf16(const float4* __restrict__ src,
                             unsigned long long* __restrict__ dst, int n4) {
  int i = blockIdx.x * blockDim.x + threadIdx.x;
  if (i < n4) {
    float4 f = src[i];
    union { bf16 b[4]; unsigned long long u; } t;
    t.b[0] = (bf16)f.x; t.b[1] = (bf16)f.y;
    t.b[2] = (bf16)f.z; t.b[3] = (bf16)f.w;
    dst[i] = t.u;
  }
}

// Load a 16x32 bf16 fragment (A layout per CDNA5 ISA 7.12.2):
// lane m = lane&15, khalf = lane>>4; two 16-byte chunks at k = khalf*8 and
// k = 16 + khalf*8. Caller passes p = &tile[row(m)][khalf*8].
__device__ __forceinline__ v16bf ld_frag(const bf16* p) {
  union { uint4 u[2]; v16bf v; } t;
  t.u[0] = *(const uint4*)(p);
  t.u[1] = *(const uint4*)(p + 16);
  return t.v;
}

// ---------------------------------------------------------------------------
// Tensor Data Mover: 2D tile (tile_d1 rows x tile_d0 elems, bf16) from global
// to LDS, with LDS row padding 32 elems -> 48-elem pitch:
//   data_size=1 (2B), pad_interval code 3 (= 16 DWORDs = one 32-elem row),
//   pad_amount code 7 (= 8 DWORDs = 16 elems).
// D# layout per CDNA5 ISA (groups 0/1; groups 2/3 unused).
// ---------------------------------------------------------------------------
__device__ __forceinline__ void tdm_load_2d_p48(unsigned lds_addr,
                                                const void* gaddr,
                                                unsigned tensor_d0,
                                                unsigned tensor_d1,
                                                unsigned long long d0_stride,
                                                unsigned tile_d0,
                                                unsigned tile_d1) {
  unsigned long long ga = (unsigned long long)(size_t)gaddr;
  u32x4 g0;
  g0[0] = 1u;                                             // count=1, user desc
  g0[1] = lds_addr;                                       // LDS byte address
  g0[2] = (unsigned)(ga & 0xFFFFFFFFu);                   // global_addr[31:0]
  g0[3] = (unsigned)((ga >> 32) & 0x01FFFFFFu)            // global_addr[56:32]
        | 0x80000000u;                                    // type=2 ("image")
  i32x8 g1;
  g1[0] = (int)(0x10000u        // data_size = 1 (2 bytes)
              | 0x100000u       // pad_enable
              | (3u << 22)      // pad_interval: 16 DWORDs
              | (7u << 25));    // pad_amount: 8 DWORDs
  g1[1] = (int)((tensor_d0 & 0xFFFFu) << 16);                    // dim0[15:0]
  g1[2] = (int)(((tensor_d0 >> 16) & 0xFFFFu)                    // dim0[31:16]
              | ((tensor_d1 & 0xFFFFu) << 16));                  // dim1[15:0]
  g1[3] = (int)(((tensor_d1 >> 16) & 0xFFFFu)                    // dim1[31:16]
              | ((tile_d0 & 0xFFFFu) << 16));                    // tile_dim0
  g1[4] = (int)(tile_d1 & 0xFFFFu);                              // tile_dim1
  g1[5] = (int)(unsigned)(d0_stride & 0xFFFFFFFFull);            // stride[31:0]
  g1[6] = (int)(unsigned)((d0_stride >> 32) & 0xFFFFull);        // stride[47:32]
  g1[7] = 0;
  i32x4 z4 = {0, 0, 0, 0};
#if __clang_major__ >= 23
  i32x8 z8 = {0, 0, 0, 0, 0, 0, 0, 0};
  __builtin_amdgcn_tensor_load_to_lds(g0, g1, z4, z4, z8, 0);
#else
  __builtin_amdgcn_tensor_load_to_lds(g0, g1, z4, z4, 0);
#endif
}

// ---------------------------------------------------------------------------
// GEMM: C[M,N] = A[M,K](bf16) * B[N,K](bf16)^T   (both K-contiguous, "NT")
// Block tile 128x128, BK=32, TDM-staged + double-buffered LDS.
// 4 waves, each a 64x64 wave tile = 16 independent WMMA chains per k-step.
// ---------------------------------------------------------------------------
template <bool F32OUT>
__global__ __launch_bounds__(128)
void gemm_nt_bf16(const bf16* __restrict__ A, const bf16* __restrict__ B,
                  float* __restrict__ Cf, bf16* __restrict__ Cb,
                  int M, int N, int K) {
  constexpr int BM = 128, BN = 128, BK = 32, LDT = 48;  // pitch set by TDM pad
  __shared__ __align__(16) bf16 As[2][BM * LDT];
  __shared__ __align__(16) bf16 Bs[2][BN * LDT];

  const int tid  = threadIdx.x;
  const int lane = tid & 31;
  const int wave = tid >> 5;   // 0..3
  const int wm   = wave >> 1;  // 0..1 -> 64-row slice
  const int wn   = wave & 1;   // 0..1 -> 64-col slice
  const int bm   = blockIdx.y * BM;
  const int bn   = blockIdx.x * BN;
  const int mlo  = lane & 15;
  const int khalf = lane >> 4;

  v8f acc[4][4];
  for (int i = 0; i < 4; ++i)
    for (int j = 0; j < 4; ++j)
      for (int e = 0; e < 8; ++e) acc[i][j][e] = 0.0f;

  const unsigned ldsA[2] = {(unsigned)(size_t)&As[0][0],
                            (unsigned)(size_t)&As[1][0]};
  const unsigned ldsB[2] = {(unsigned)(size_t)&Bs[0][0],
                            (unsigned)(size_t)&Bs[1][0]};

  // Prologue: DMA tile 0 into buffer 0 (one wave drives the TDM).
  if (wave == 0) {
    tdm_load_2d_p48(ldsA[0], A + (size_t)bm * K, (unsigned)K, (unsigned)M,
                    (unsigned long long)K, BK, BM);
    tdm_load_2d_p48(ldsB[0], B + (size_t)bn * K, (unsigned)K, (unsigned)N,
                    (unsigned long long)K, BK, BN);
  }

  const int NT = K / BK;
  for (int t = 0; t < NT; ++t) {
    const int cur = t & 1;
    if (wave == 0) {
      if (t + 1 < NT) {
        // Issue next tile into the other buffer, then wait for current tile
        // (<=2 outstanding: the two just-issued TDM ops keep flying).
        const size_t ko = (size_t)(t + 1) * BK;
        tdm_load_2d_p48(ldsA[cur ^ 1], A + (size_t)bm * K + ko, (unsigned)K,
                        (unsigned)M, (unsigned long long)K, BK, BM);
        tdm_load_2d_p48(ldsB[cur ^ 1], B + (size_t)bn * K + ko, (unsigned)K,
                        (unsigned)N, (unsigned long long)K, BK, BN);
        __builtin_amdgcn_s_wait_tensorcnt(2);
      } else {
        __builtin_amdgcn_s_wait_tensorcnt(0);
      }
    }
    __syncthreads();

    const bf16* at = As[cur];
    const bf16* bt = Bs[cur];
    v16bf af[4], bfr[4];
    for (int i = 0; i < 4; ++i)
      af[i] = ld_frag(&at[(wm * 64 + i * 16 + mlo) * LDT + khalf * 8]);
    for (int j = 0; j < 4; ++j)
      bfr[j] = ld_frag(&bt[(wn * 64 + j * 16 + mlo) * LDT + khalf * 8]);

    for (int i = 0; i < 4; ++i)
      for (int j = 0; j < 4; ++j)
        acc[i][j] = __builtin_amdgcn_wmma_f32_16x16x32_bf16(
            false, af[i], false, bfr[j], (short)0, acc[i][j], false, false);
    __syncthreads();
  }

  // Epilogue: C layout -> VGPR r holds row (r + 8*khalf), col = mlo
  for (int i = 0; i < 4; ++i)
    for (int j = 0; j < 4; ++j) {
      int rbase = bm + wm * 64 + i * 16 + khalf * 8;
      int c     = bn + wn * 64 + j * 16 + mlo;
      for (int r = 0; r < 8; ++r) {
        float v = acc[i][j][r];
        size_t off = (size_t)(rbase + r) * N + c;
        if (F32OUT) Cf[off] = v;
        else        Cb[off] = (bf16)v;
      }
    }
}

// ---------------------------------------------------------------------------
// Flash attention: block = 64 q-rows of one (batch, head); 4 waves x 16 rows.
// Loop over 64-key tiles: S = Q K^T (WMMA), online softmax, P V (WMMA).
// ---------------------------------------------------------------------------
__global__ __launch_bounds__(128)
void flash_attn(const bf16* __restrict__ Q, const bf16* __restrict__ Km,
                const bf16* __restrict__ Vm, bf16* __restrict__ O) {
  constexpr int LDK = 136;  // K-tile LDS row stride (272B, 16B aligned)
  constexpr int LDV = 72;   // transposed-V LDS row stride (144B, 16B aligned)
  constexpr int LDP = 72;   // P scratch row stride
  __shared__ __align__(16) bf16 Ks[64 * LDK];       // 64 keys x 128 hd
  __shared__ __align__(16) bf16 Vt[HDIM * LDV];     // 128 hd  x 64 keys
  __shared__ __align__(16) bf16 Ps[4 * 16 * LDP];   // per-wave 16 x 64 P tile

  const int tid   = threadIdx.x;
  const int lane  = tid & 31;
  const int wave  = tid >> 5;
  const int mlo   = lane & 15;
  const int khalf = lane >> 4;
  const int bh = blockIdx.y;
  const int b  = bh >> 4;
  const int h  = bh & 15;
  const int q0 = blockIdx.x * 64 + wave * 16;
  const size_t base = (size_t)b * SEQ * EMB + (size_t)h * HDIM;
  const float scale = 0.08838834764831845f;  // 1/sqrt(128)

  // Q fragments (scale folded in): 4 x (16x32) covering hd = 128
  v16bf aq[4];
  for (int d = 0; d < 4; ++d) {
    const bf16* p = Q + base + (size_t)(q0 + mlo) * EMB + d * 32 + khalf * 8;
    union { uint4 u[2]; v16bf v; } t;
    t.u[0] = *(const uint4*)p;
    t.u[1] = *(const uint4*)(p + 16);
    for (int e = 0; e < 16; ++e) t.v[e] = (bf16)((float)t.v[e] * scale);
    aq[d] = t.v;
  }

  float mrow[8], lrow[8];
  v8f o[8];
  for (int j = 0; j < 8; ++j) { mrow[j] = -1e30f; lrow[j] = 0.0f; }
  for (int j = 0; j < 8; ++j)
    for (int e = 0; e < 8; ++e) o[j][e] = 0.0f;

  for (int kt = 0; kt < SEQ; kt += 64) {
    // Stage K tile (as-is) and V tile (transposed): 64x128 each,
    // 1024 8-elem chunks, 128 threads -> 8 chunks per thread.
    for (int it = 0; it < 8; ++it) {
      int idx = tid + it * 128;
      int row = idx >> 4;          // key 0..63
      int col = (idx & 15) * 8;    // hd  0..120
      uint4 dk = *(const uint4*)(Km + base + (size_t)(kt + row) * EMB + col);
      *(uint4*)(&Ks[row * LDK + col]) = dk;
      union { uint4 u; bf16 e[8]; } tv;
      tv.u = *(const uint4*)(Vm + base + (size_t)(kt + row) * EMB + col);
      for (int e = 0; e < 8; ++e) Vt[(col + e) * LDV + row] = tv.e[e];
      if (kt + 64 < SEQ) {  // global_prefetch_b8 for the next tile
        __builtin_prefetch(Km + base + (size_t)(kt + 64 + row) * EMB + col, 0, 1);
        __builtin_prefetch(Vm + base + (size_t)(kt + 64 + row) * EMB + col, 0, 1);
      }
    }
    __syncthreads();

    // S = Q K^T : d-outer / n-inner so each load batch feeds 4 INDEPENDENT
    // accumulator chains (s[0..3]) -> DS latency overlaps with WMMA issue.
    v8f s[4];
    for (int n = 0; n < 4; ++n)
      for (int e = 0; e < 8; ++e) s[n][e] = 0.0f;
    for (int d = 0; d < 4; ++d) {
      const bf16* kr = &Ks[mlo * LDK + d * 32 + khalf * 8];
      v16bf bk0 = ld_frag(kr);
      v16bf bk1 = ld_frag(kr + 16 * LDK);
      v16bf bk2 = ld_frag(kr + 32 * LDK);
      v16bf bk3 = ld_frag(kr + 48 * LDK);
      s[0] = __builtin_amdgcn_wmma_f32_16x16x32_bf16(false, aq[d], false, bk0,
                                                     (short)0, s[0], false, false);
      s[1] = __builtin_amdgcn_wmma_f32_16x16x32_bf16(false, aq[d], false, bk1,
                                                     (short)0, s[1], false, false);
      s[2] = __builtin_amdgcn_wmma_f32_16x16x32_bf16(false, aq[d], false, bk2,
                                                     (short)0, s[2], false, false);
      s[3] = __builtin_amdgcn_wmma_f32_16x16x32_bf16(false, aq[d], false, bk3,
                                                     (short)0, s[3], false, false);
    }

    // Online softmax. Row (j + 8*khalf) lives in one 16-lane half.
    for (int j = 0; j < 8; ++j) {
      float loc = fmaxf(fmaxf(s[0][j], s[1][j]), fmaxf(s[2][j], s[3][j]));
      for (int msk = 8; msk >= 1; msk >>= 1)
        loc = fmaxf(loc, __shfl_xor(loc, msk, 32));
      float mn = fmaxf(mrow[j], loc);
      float alpha = __expf(mrow[j] - mn);
      float rs = 0.0f;
      for (int n = 0; n < 4; ++n) {
        float p = __expf(s[n][j] - mn);
        s[n][j] = p;
        rs += p;
      }
      for (int msk = 8; msk >= 1; msk >>= 1)
        rs += __shfl_xor(rs, msk, 32);
      lrow[j] = lrow[j] * alpha + rs;
      mrow[j] = mn;
      for (int nd = 0; nd < 8; ++nd) o[nd][j] *= alpha;
    }

    // P: C-layout -> LDS -> A-layout
    bf16* pw = &Ps[wave * 16 * LDP];
    for (int n = 0; n < 4; ++n)
      for (int j = 0; j < 8; ++j)
        pw[(j + khalf * 8) * LDP + n * 16 + mlo] = (bf16)s[n][j];
    __syncthreads();

    // O += P V : 2 key chunks x (2 groups of 4 independent hd chains)
    for (int kf = 0; kf < 2; ++kf) {
      v16bf ap = ld_frag(&pw[mlo * LDP + kf * 32 + khalf * 8]);
      for (int g = 0; g < 2; ++g) {
        const bf16* vr = &Vt[(g * 64 + mlo) * LDV + kf * 32 + khalf * 8];
        v16bf bv0 = ld_frag(vr);
        v16bf bv1 = ld_frag(vr + 16 * LDV);
        v16bf bv2 = ld_frag(vr + 32 * LDV);
        v16bf bv3 = ld_frag(vr + 48 * LDV);
        o[g * 4 + 0] = __builtin_amdgcn_wmma_f32_16x16x32_bf16(
            false, ap, false, bv0, (short)0, o[g * 4 + 0], false, false);
        o[g * 4 + 1] = __builtin_amdgcn_wmma_f32_16x16x32_bf16(
            false, ap, false, bv1, (short)0, o[g * 4 + 1], false, false);
        o[g * 4 + 2] = __builtin_amdgcn_wmma_f32_16x16x32_bf16(
            false, ap, false, bv2, (short)0, o[g * 4 + 2], false, false);
        o[g * 4 + 3] = __builtin_amdgcn_wmma_f32_16x16x32_bf16(
            false, ap, false, bv3, (short)0, o[g * 4 + 3], false, false);
      }
    }
    __syncthreads();
  }

  // Normalize and store (bf16, same (B,S,E) layout)
  for (int nd = 0; nd < 8; ++nd)
    for (int j = 0; j < 8; ++j) {
      float val = o[nd][j] / lrow[j];
      size_t off = base + (size_t)(q0 + j + khalf * 8) * EMB + nd * 16 + mlo;
      O[off] = (bf16)val;
    }
}

// ---------------------------------------------------------------------------
// Host-side orchestration
// ---------------------------------------------------------------------------
extern "C" void kernel_launch(void* const* d_in, const int* in_sizes, int n_in,
                              void* d_out, int out_size, void* d_ws, size_t ws_size,
                              hipStream_t stream) {
  const float* x  = (const float*)d_in[0];
  const float* wq = (const float*)d_in[1];
  const float* wk = (const float*)d_in[2];
  const float* wv = (const float*)d_in[3];
  const float* wo = (const float*)d_in[4];

  const size_t nX = (size_t)BATCH * SEQ * EMB;  // 4096*2048
  const size_t nW = (size_t)EMB * EMB;          // 2048*2048

  bf16* xb  = (bf16*)d_ws;
  bf16* wqb = xb  + nX;
  bf16* wkb = wqb + nW;
  bf16* wvb = wkb + nW;
  bf16* wob = wvb + nW;
  bf16* qb  = wob + nW;
  bf16* kb  = qb  + nX;
  bf16* vb  = kb  + nX;
  bf16* ob  = vb  + nX;   // total ~117 MB of bf16 scratch

  const int nX4 = (int)(nX / 4), nW4 = (int)(nW / 4);
  cvt_f32_bf16<<<(nX4 + 255) / 256, 256, 0, stream>>>(
      (const float4*)x, (unsigned long long*)xb, nX4);
  cvt_f32_bf16<<<(nW4 + 255) / 256, 256, 0, stream>>>(
      (const float4*)wq, (unsigned long long*)wqb, nW4);
  cvt_f32_bf16<<<(nW4 + 255) / 256, 256, 0, stream>>>(
      (const float4*)wk, (unsigned long long*)wkb, nW4);
  cvt_f32_bf16<<<(nW4 + 255) / 256, 256, 0, stream>>>(
      (const float4*)wv, (unsigned long long*)wvb, nW4);
  cvt_f32_bf16<<<(nW4 + 255) / 256, 256, 0, stream>>>(
      (const float4*)wo, (unsigned long long*)wob, nW4);

  const int M = BATCH * SEQ, N = EMB, K = EMB;
  dim3 gg(N / 128, M / 128);   // 16 x 32 blocks, 128 threads each
  gemm_nt_bf16<false><<<gg, 128, 0, stream>>>(xb, wqb, nullptr, qb, M, N, K);
  gemm_nt_bf16<false><<<gg, 128, 0, stream>>>(xb, wkb, nullptr, kb, M, N, K);
  gemm_nt_bf16<false><<<gg, 128, 0, stream>>>(xb, wvb, nullptr, vb, M, N, K);

  dim3 ga(SEQ / 64, BATCH * HEADS);  // 32 q-tiles x 32 (b,h)
  flash_attn<<<ga, 128, 0, stream>>>(qb, kb, vb, ob);

  gemm_nt_bf16<true><<<gg, 128, 0, stream>>>(ob, wob, (float*)d_out, nullptr, M, N, K);
}
